// VectorQuantizerFC_48430051230510
// MI455X (gfx1250) — matrix operand order repaired and checked by reading, not compiled, not run on hardware
//
#include <hip/hip_runtime.h>
#include <stdint.h>

// ---------------------------------------------------------------------------
// VQ soft-quantizer for gfx1250 (MI455X).
//   out = [ z_q (B*256 f32) | P (B*512 f32) | loss (1 f32) ]
// GEMM1: G = z @ W^T (bf16 WMMA, f32 acc), logits = (2G - ||w||^2)*100
//        (||z||^2 is constant per row -> cancels in softmax)
// softmax over N=512, GEMM2: z_q = P @ W (bf16 WMMA), loss = 1.25*mse.
// Codebook operands are staged LDS-side with double-buffered
// global_load_async_to_lds_b128 (ASYNCcnt) so WMMA is fed from LDS.
// ---------------------------------------------------------------------------

#define NUM_EMB   512
#define EMB_DIM   256
#define BATCH     262144
#define INV_T     100.0f
#define ROWS_PER_WG 64            // 4 M-tiles, 2 waves per tile
#define LDS_P_STRIDE 520          // halves; 1040B rows -> conflict-free b128

// LDS layout (bytes, within dynamic alloc)
#define CHUNK_ROW_B   80          // 64B data + 16B pad -> conflict-free b128
#define CHUNK_BYTES   (512 * CHUNK_ROW_B)          // 40960
#define CHUNK0_OFF    0
#define CHUNK1_OFF    CHUNK_BYTES
#define P_OFF         (2 * CHUNK_BYTES)            // 81920
#define P_TILE_BYTES  (16 * LDS_P_STRIDE * 2)      // 16640
#define REDMAX_OFF    (P_OFF + 4 * P_TILE_BYTES)   // 148480
#define REDSUM_OFF    (REDMAX_OFF + 512)
#define SMEM_BYTES    (REDSUM_OFF + 512)           // 149504

typedef __attribute__((ext_vector_type(16))) __bf16 v16bf;
typedef __attribute__((ext_vector_type(8)))  float  v8f;

union Frag16 {
  v16bf v;
  uint4 q[2];
};

__device__ __forceinline__ uint16_t bf16_1(float a) {
  uint32_t u = __float_as_uint(a);
  return (uint16_t)((u + 0x7FFFu + ((u >> 16) & 1u)) >> 16);
}

// gfx1250 async global->LDS DMA (ISA §15.18.3 op 98). VDST VGPR holds the
// wave-relative LDS byte address == low 32 bits of the generic shared pointer.
__device__ __forceinline__ void async_ld_b128(uint32_t lds_off, const void* g) {
  asm volatile("global_load_async_to_lds_b128 %0, %1, off"
               :: "v"(lds_off), "v"((uint64_t)(uintptr_t)g)
               : "memory");
}
__device__ __forceinline__ void wait_async0() {
  asm volatile("s_wait_asynccnt 0x0" ::: "memory");
}

// Stage NROWS codebook rows x 32 halves (one K-chunk) into LDS, padded rows.
template <int NROWS>
__device__ __forceinline__ void issue_chunk(const uint16_t* gbase,
                                            int rowStrideHalves, int kOffHalves,
                                            uint32_t ldsBase, int tid) {
#pragma unroll
  for (int i = 0; i < NROWS * 4 / 256; ++i) {
    int idx  = i * 256 + tid;
    int n    = idx >> 2;
    int part = idx & 3;
    async_ld_b128(ldsBase + (uint32_t)(n * CHUNK_ROW_B + part * 16),
                  gbase + (size_t)n * rowStrideHalves + kOffHalves + part * 8);
  }
}

// ---------------------------------------------------------------------------
// Prep: bf16 codebook (Wb row-major for GEMM1-B, Wt transposed for GEMM2-B),
// squared norms, zero loss. 512 blocks x 256 threads.
// ---------------------------------------------------------------------------
__global__ void vq_prep_kernel(const float* __restrict__ W,
                               uint16_t* __restrict__ Wb,
                               uint16_t* __restrict__ Wt,
                               float* __restrict__ wnorm,
                               float* __restrict__ loss_slot) {
  const int n = blockIdx.x;
  const int d = threadIdx.x;
  float w = W[n * EMB_DIM + d];
  uint16_t h = bf16_1(w);
  Wb[n * EMB_DIM + d] = h;
  Wt[d * NUM_EMB + n] = h;
  float p = w * w;
#pragma unroll
  for (int m = 16; m >= 1; m >>= 1) p += __shfl_xor(p, m, 32);
  __shared__ float red[8];
  if ((threadIdx.x & 31) == 0) red[threadIdx.x >> 5] = p;
  __syncthreads();
  if (threadIdx.x == 0) {
    float s = 0.f;
#pragma unroll
    for (int i = 0; i < 8; ++i) s += red[i];
    wnorm[n] = s;
    if (n == 0) *loss_slot = 0.0f;
  }
}

// ---------------------------------------------------------------------------
// Main kernel: 256 threads, 8 waves; wave pair (tile, nh) shares a 16-row tile.
// ---------------------------------------------------------------------------
__global__ void __launch_bounds__(256)
vq_main_kernel(const float* __restrict__ z_e,
               const uint16_t* __restrict__ Wb,   // [512][256] bf16
               const uint16_t* __restrict__ Wt,   // [256][512] bf16
               const float* __restrict__ wnorm,   // [512]
               float* __restrict__ out_zq,
               float* __restrict__ out_p,
               float* __restrict__ loss) {
  extern __shared__ uint8_t smem[];
  const uint32_t smemLow = (uint32_t)(uintptr_t)smem;  // LDS byte base
  const int tid  = threadIdx.x;
  const int wave = tid >> 5;
  const int lane = tid & 31;
  const int li   = lane & 15;
  const int hi   = lane >> 4;
  const int tile = wave >> 1;        // 0..3 -> 16-row M-tile
  const int nh   = wave & 1;         // N-half (GEMM1) / D-half (GEMM2)
  const int rowBase = blockIdx.x * ROWS_PER_WG + tile * 16;

  __bf16* plds = (__bf16*)(smem + P_OFF) + tile * 16 * LDS_P_STRIDE;
  float*  redmax = (float*)(smem + REDMAX_OFF);     // [4][2][16]
  float*  redsum = (float*)(smem + REDSUM_OFF);

  // ================= GEMM1: G = z @ W^T (this wave: 16 n-tiles) ===========
  v8f acc[16];
#pragma unroll
  for (int nt = 0; nt < 16; ++nt)
#pragma unroll
    for (int v = 0; v < 8; ++v) acc[nt][v] = 0.0f;

  const float* zrow = z_e + (size_t)(rowBase + li) * EMB_DIM;

  // A-fragment loader: lanes 0-15 K(0..7,16..23), lanes 16-31 K(8..15,24..31)
  auto loadA = [&](int kk) {
    Frag16 a;
    const float4* f0 = (const float4*)(zrow + kk * 32 + hi * 8);
    const float4* f1 = (const float4*)(zrow + kk * 32 + 16 + hi * 8);
    float4 x0 = f0[0], x1 = f0[1], x2 = f1[0], x3 = f1[1];
    a.v[0]  = (__bf16)x0.x; a.v[1]  = (__bf16)x0.y;
    a.v[2]  = (__bf16)x0.z; a.v[3]  = (__bf16)x0.w;
    a.v[4]  = (__bf16)x1.x; a.v[5]  = (__bf16)x1.y;
    a.v[6]  = (__bf16)x1.z; a.v[7]  = (__bf16)x1.w;
    a.v[8]  = (__bf16)x2.x; a.v[9]  = (__bf16)x2.y;
    a.v[10] = (__bf16)x2.z; a.v[11] = (__bf16)x2.w;
    a.v[12] = (__bf16)x3.x; a.v[13] = (__bf16)x3.y;
    a.v[14] = (__bf16)x3.z; a.v[15] = (__bf16)x3.w;
    return a;
  };

  // prologue: DMA codebook K-chunk 0, overlap with first A-fragment load
  issue_chunk<512>(Wb, EMB_DIM, 0, smemLow + CHUNK0_OFF, tid);
  Frag16 a = loadA(0);
  wait_async0();
  __syncthreads();

  for (int kk = 0; kk < 8; ++kk) {
    if (kk < 7)   // DMA next chunk while computing this one
      issue_chunk<512>(Wb, EMB_DIM, (kk + 1) * 32,
                       smemLow + ((kk + 1) & 1 ? CHUNK1_OFF : CHUNK0_OFF), tid);
    Frag16 anext = (kk < 7) ? loadA(kk + 1) : a;
    const uint8_t* buf = smem + ((kk & 1) ? CHUNK1_OFF : CHUNK0_OFF);
#pragma unroll
    for (int nt = 0; nt < 16; ++nt) {
      int ncol = (nh * 16 + nt) * 16 + li;         // codebook entry (B col)
      Frag16 b;
      const uint4* pb = (const uint4*)(buf + ncol * CHUNK_ROW_B + hi * 16);
      b.q[0] = pb[0]; b.q[1] = pb[2];              // ds_load_b128 x2
      acc[nt] = __builtin_amdgcn_wmma_f32_16x16x32_bf16(
          false, a.v, false, b.v, (short)0, acc[nt], false, false);
    }
    wait_async0();
    __syncthreads();
    a = anext;
  }

  // ---------------- logits ------------------------------------------------
#pragma unroll
  for (int nt = 0; nt < 16; ++nt) {
    float wn = wnorm[(nh * 16 + nt) * 16 + li];
#pragma unroll
    for (int v = 0; v < 8; ++v)
      acc[nt][v] = (2.0f * acc[nt][v] - wn) * INV_T;
  }

  // ---------------- softmax across the wave pair --------------------------
  // C/D layout: element(lane, v) = row (v + 8*hi), col (lane%16 + 16*nt)
  float mrow[8], inv[8];
#pragma unroll
  for (int v = 0; v < 8; ++v) {
    float t = acc[0][v];
#pragma unroll
    for (int nt = 1; nt < 16; ++nt) t = fmaxf(t, acc[nt][v]);
    t = fmaxf(t, __shfl_xor(t, 1, 32));
    t = fmaxf(t, __shfl_xor(t, 2, 32));
    t = fmaxf(t, __shfl_xor(t, 4, 32));
    t = fmaxf(t, __shfl_xor(t, 8, 32));
    mrow[v] = t;
  }
  if (li == 0) {
#pragma unroll
    for (int v = 0; v < 8; ++v)
      redmax[(tile * 2 + nh) * 16 + v + 8 * hi] = mrow[v];
  }
  __syncthreads();
#pragma unroll
  for (int v = 0; v < 8; ++v)
    mrow[v] = fmaxf(mrow[v], redmax[(tile * 2 + (1 - nh)) * 16 + v + 8 * hi]);

  float srow[8];
#pragma unroll
  for (int v = 0; v < 8; ++v) {
    float s = 0.f;
#pragma unroll
    for (int nt = 0; nt < 16; ++nt) {
      float e = __expf(acc[nt][v] - mrow[v]);
      acc[nt][v] = e;
      s += e;
    }
    s += __shfl_xor(s, 1, 32);
    s += __shfl_xor(s, 2, 32);
    s += __shfl_xor(s, 4, 32);
    s += __shfl_xor(s, 8, 32);
    srow[v] = s;
  }
  if (li == 0) {
#pragma unroll
    for (int v = 0; v < 8; ++v)
      redsum[(tile * 2 + nh) * 16 + v + 8 * hi] = srow[v];
  }
  __syncthreads();
#pragma unroll
  for (int v = 0; v < 8; ++v)
    inv[v] = 1.0f / (srow[v] + redsum[(tile * 2 + (1 - nh)) * 16 + v + 8 * hi]);

  // -------- P: f32 non-temporal to global + bf16 staging in LDS -----------
#pragma unroll
  for (int nt = 0; nt < 16; ++nt) {
#pragma unroll
    for (int v = 0; v < 8; ++v) {
      float pval = acc[nt][v] * inv[v];
      int r   = v + 8 * hi;
      int col = (nh * 16 + nt) * 16 + li;
      __builtin_nontemporal_store(
          pval, out_p + (size_t)(rowBase + r) * NUM_EMB + col);
      plds[r * LDS_P_STRIDE + col] = (__bf16)pval;
    }
  }

  // ================= GEMM2: z_q = P @ W (this wave: 8 d-tiles) ============
  v8f acc2[8];
#pragma unroll
  for (int nt = 0; nt < 8; ++nt)
#pragma unroll
    for (int v = 0; v < 8; ++v) acc2[nt][v] = 0.0f;

  // DMA Wt K-chunk 0 (overlaps the barrier that publishes P in LDS)
  issue_chunk<256>(Wt, NUM_EMB, 0, smemLow + CHUNK0_OFF, tid);
  wait_async0();
  __syncthreads();                                  // P visible + chunk0 ready

  const __bf16* prow = plds + li * LDS_P_STRIDE;    // A row for this lane
  for (int kt = 0; kt < 16; ++kt) {
    if (kt < 15)
      issue_chunk<256>(Wt, NUM_EMB, (kt + 1) * 32,
                       smemLow + ((kt + 1) & 1 ? CHUNK1_OFF : CHUNK0_OFF), tid);
    Frag16 a2;
    const uint4* pa0 = (const uint4*)(prow + kt * 32 + hi * 8);
    const uint4* pa1 = (const uint4*)(prow + kt * 32 + 16 + hi * 8);
    a2.q[0] = pa0[0];
    a2.q[1] = pa1[0];
    const uint8_t* buf = smem + ((kt & 1) ? CHUNK1_OFF : CHUNK0_OFF);
#pragma unroll
    for (int nt = 0; nt < 8; ++nt) {
      int d = (nh * 8 + nt) * 16 + li;              // output dim (B col)
      Frag16 b;
      const uint4* pb = (const uint4*)(buf + d * CHUNK_ROW_B + hi * 16);
      b.q[0] = pb[0]; b.q[1] = pb[2];
      acc2[nt] = __builtin_amdgcn_wmma_f32_16x16x32_bf16(
          false, a2.v, false, b.v, (short)0, acc2[nt], false, false);
    }
    wait_async0();
    __syncthreads();
  }

  // ---------------- write z_q (== z_q_st) + loss --------------------------
  float d2 = 0.f;
#pragma unroll
  for (int nt = 0; nt < 8; ++nt) {
#pragma unroll
    for (int v = 0; v < 8; ++v) {
      int row = rowBase + v + 8 * hi;
      int col = (nh * 8 + nt) * 16 + li;
      size_t idx = (size_t)row * EMB_DIM + col;
      float zq = acc2[nt][v];
      float ze = z_e[idx];
      __builtin_nontemporal_store(zq, out_zq + idx);
      float df = zq - ze;
      d2 = fmaf(df, df, d2);
    }
  }
  d2 += __shfl_xor(d2, 1, 32);
  d2 += __shfl_xor(d2, 2, 32);
  d2 += __shfl_xor(d2, 4, 32);
  d2 += __shfl_xor(d2, 8, 32);
  d2 += __shfl_xor(d2, 16, 32);
  if (lane == 0) atomicAdd(loss, d2 * (1.25f / 67108864.0f));
}

// ---------------------------------------------------------------------------
extern "C" void kernel_launch(void* const* d_in, const int* in_sizes, int n_in,
                              void* d_out, int out_size, void* d_ws,
                              size_t ws_size, hipStream_t stream) {
  (void)in_sizes; (void)n_in; (void)out_size; (void)ws_size;
  const float* z_e = (const float*)d_in[0];
  const float* W   = (const float*)d_in[1];

  float* out    = (float*)d_out;
  float* out_zq = out;
  float* out_p  = out + (size_t)BATCH * EMB_DIM;
  float* loss   = out + (size_t)BATCH * EMB_DIM + (size_t)BATCH * NUM_EMB;

  uint8_t* ws = (uint8_t*)d_ws;
  uint16_t* Wb    = (uint16_t*)(ws);
  uint16_t* Wt    = (uint16_t*)(ws + 262144);
  float*    wnorm = (float*)(ws + 524288);

  vq_prep_kernel<<<NUM_EMB, EMB_DIM, 0, stream>>>(W, Wb, Wt, wnorm, loss);
  vq_main_kernel<<<BATCH / ROWS_PER_WG, 256, SMEM_BYTES, stream>>>(
      z_e, Wb, Wt, wnorm, out_zq, out_p, loss);
}
